// Factorized_SoftmaxV2_4028679324208
// MI455X (gfx1250) — matrix operand order repaired and checked by reading, not compiled, not run on hardware
//
#include <hip/hip_runtime.h>

typedef __attribute__((ext_vector_type(16))) __bf16 v16bf;
typedef __attribute__((ext_vector_type(8)))  __bf16 v8bf;
typedef __attribute__((ext_vector_type(8)))  float  v8f;

#define NTOK 4096
#define HID  1024
#define VOC  50257
#define PITCH 1032               // bf16 elems per LDS column: 16B-aligned, bank-spread
#define NBLK ((VOC + 63) / 64)   // 786 vocab tiles of 64 columns

__device__ __forceinline__ int cluster_of(int col) {
  return (col < 2000) ? 0 : (col < 10000) ? 1 : (col < 30000) ? 2 : 3;
}

// Batched 16-lane-half butterfly reduction: 8 independent rows per round so the
// compiler can clause the ds_bpermutes and use a single dscnt wait per round.
__device__ __forceinline__ void reduce_half16(float (&v)[8]) {
  #pragma unroll
  for (int mask = 1; mask <= 8; mask <<= 1) {
    float t[8];
    #pragma unroll
    for (int g = 0; g < 8; ++g) t[g] = __shfl_xor(v[g], mask);
    #pragma unroll
    for (int g = 0; g < 8; ++g) v[g] += t[g];
  }
}

// ---------------------------------------------------------------- prep
__global__ void prep_kernel(const float* __restrict__ x, __bf16* __restrict__ xb,
                            float* __restrict__ sumexp) {
  int i = blockIdx.x * blockDim.x + threadIdx.x;
  if (i < NTOK * HID) xb[i] = (__bf16)x[i];
  if (i < NTOK * 4)   sumexp[i] = 0.0f;
}

// ---------------------------------------------------------------- fused GEMM + exp-sum
// 4 waves, 2x2 wave grid; each wave owns a 32x32 output tile (2x2 WMMA tiles
// with register reuse: 1 global b128 + 1 ds b128 per v_wmma).
__global__ __launch_bounds__(128) void gemm_kernel(const __bf16* __restrict__ xb,
                                                   const float* __restrict__ W,
                                                   float* __restrict__ sumexp) {
  __shared__ __bf16 wlds[64 * PITCH];   // 64 cols x 1024 K (bf16), column-major

  const int tid    = threadIdx.x;
  const int col_wg = blockIdx.x * 64;

  // Stage W tile fp32 -> bf16 into LDS (coalesced global reads).
  for (int idx = tid; idx < 64 * HID; idx += 128) {
    int c = idx & 63;
    int k = idx >> 6;
    int col = col_wg + c;
    float w = (col < VOC) ? W[(size_t)k * VOC + col] : 0.0f;
    wlds[c * PITCH + k] = (__bf16)w;
  }
  __syncthreads();

  const int  wave = tid >> 5;
  const int  lane = tid & 31;
  const int  lrow = lane & 15;
  const bool hi   = lane >= 16;
  const int  nq   = wave & 1;          // wave col-quadrant (2 x 32 cols)
  const int  mq   = wave >> 1;         // wave row-quadrant (2 x 32 rows)

  const int  c0 = col_wg + nq * 32;          // n-subtile 0 base
  const int  c1 = c0 + 16;                   // n-subtile 1 base
  const bool valid0 = (c0 + lrow) < VOC;
  const bool valid1 = (c1 + lrow) < VOC;
  const int  cid0 = cluster_of(c0);
  const int  cid1 = cluster_of(c1);

  // B (32x16 bf16): lane L (0-15) holds col N=L with K{0..7,16..23}, lane L+16 K{8..15,24..31}.
  const __bf16* bcol0 = &wlds[(size_t)(nq * 32 + lrow) * PITCH + (hi ? 8 : 0)];
  const __bf16* bcol1 = bcol0 + 16 * PITCH;

  for (int m0 = 0; m0 < NTOK; m0 += 64) {
    // A (16x32 bf16): lanes 0-15 row=lrow K{0..7,16..23}; lanes 16-31 K{8..15,24..31}
    const __bf16* arow0 = xb + (size_t)(m0 + mq * 32 + lrow) * HID + (hi ? 8 : 0);
    const __bf16* arow1 = arow0 + 16 * HID;
    v8f acc00 = {}, acc01 = {}, acc10 = {}, acc11 = {};
    #pragma unroll 2
    for (int k0 = 0; k0 < HID; k0 += 32) {
      v8bf a0l = *(const v8bf*)(arow0 + k0);
      v8bf a0h = *(const v8bf*)(arow0 + k0 + 16);
      v8bf a1l = *(const v8bf*)(arow1 + k0);
      v8bf a1h = *(const v8bf*)(arow1 + k0 + 16);
      v8bf b0l = *(const v8bf*)(bcol0 + k0);
      v8bf b0h = *(const v8bf*)(bcol0 + k0 + 16);
      v8bf b1l = *(const v8bf*)(bcol1 + k0);
      v8bf b1h = *(const v8bf*)(bcol1 + k0 + 16);
      v16bf a0 = __builtin_shufflevector(a0l, a0h, 0,1,2,3,4,5,6,7,8,9,10,11,12,13,14,15);
      v16bf a1 = __builtin_shufflevector(a1l, a1h, 0,1,2,3,4,5,6,7,8,9,10,11,12,13,14,15);
      v16bf b0 = __builtin_shufflevector(b0l, b0h, 0,1,2,3,4,5,6,7,8,9,10,11,12,13,14,15);
      v16bf b1 = __builtin_shufflevector(b1l, b1h, 0,1,2,3,4,5,6,7,8,9,10,11,12,13,14,15);
      acc00 = __builtin_amdgcn_wmma_f32_16x16x32_bf16(false, a0, false, b0, (short)0, acc00, false, false);
      acc01 = __builtin_amdgcn_wmma_f32_16x16x32_bf16(false, a0, false, b1, (short)0, acc01, false, false);
      acc10 = __builtin_amdgcn_wmma_f32_16x16x32_bf16(false, a1, false, b0, (short)0, acc10, false, false);
      acc11 = __builtin_amdgcn_wmma_f32_16x16x32_bf16(false, a1, false, b1, (short)0, acc11, false, false);
    }

    // Epilogue per 16-row m-subtile: exp, (maybe fused) cross-lane reduce, atomics.
    #pragma unroll
    for (int mt = 0; mt < 2; ++mt) {
      const v8f& an0 = mt ? acc10 : acc00;
      const v8f& an1 = mt ? acc11 : acc01;
      const int rowbase = m0 + mq * 32 + mt * 16 + (hi ? 8 : 0);
      float e0[8], e1[8];
      #pragma unroll
      for (int g = 0; g < 8; ++g) e0[g] = valid0 ? __expf(an0[g]) : 0.0f;
      #pragma unroll
      for (int g = 0; g < 8; ++g) e1[g] = valid1 ? __expf(an1[g]) : 0.0f;
      if (cid0 == cid1) {                 // wave-uniform: fuse before reduction
        #pragma unroll
        for (int g = 0; g < 8; ++g) e0[g] += e1[g];
        reduce_half16(e0);
        if (lrow == 0) {
          #pragma unroll
          for (int g = 0; g < 8; ++g)
            atomicAdd(&sumexp[(rowbase + g) * 4 + cid0], e0[g]);
        }
      } else {
        reduce_half16(e0);
        reduce_half16(e1);
        if (lrow == 0) {
          #pragma unroll
          for (int g = 0; g < 8; ++g) {
            atomicAdd(&sumexp[(rowbase + g) * 4 + cid0], e0[g]);
            atomicAdd(&sumexp[(rowbase + g) * 4 + cid1], e1[g]);
          }
        }
      }
    }
  }
}

// ---------------------------------------------------------------- exact fp32 target logit
__global__ __launch_bounds__(128) void tgt_kernel(const float* __restrict__ x,
                                                  const int* __restrict__ y,
                                                  const float* __restrict__ W,
                                                  float* __restrict__ tgt) {
  __shared__ float red[4];
  int n   = blockIdx.x;
  int col = y[n];
  const float* xr = x + (size_t)n * HID;
  float s = 0.0f;
  for (int k = threadIdx.x; k < HID; k += 128)
    s += xr[k] * W[(size_t)k * VOC + col];
  s += __shfl_xor(s, 1);
  s += __shfl_xor(s, 2);
  s += __shfl_xor(s, 4);
  s += __shfl_xor(s, 8);
  s += __shfl_xor(s, 16);
  if ((threadIdx.x & 31) == 0) red[threadIdx.x >> 5] = s;
  __syncthreads();
  if (threadIdx.x == 0) tgt[n] = red[0] + red[1] + red[2] + red[3];
}

// ---------------------------------------------------------------- cluster head + combine
__global__ void final_kernel(const float* __restrict__ x, const int* __restrict__ y,
                             const float* __restrict__ cw, const float* __restrict__ sumexp,
                             const float* __restrict__ tgt, float* __restrict__ nll) {
  int n = blockIdx.x * blockDim.x + threadIdx.x;
  if (n >= NTOK) return;
  const float* xr = x + (size_t)n * HID;
  float cl0 = 0.f, cl1 = 0.f, cl2 = 0.f, cl3 = 0.f;
  for (int k = 0; k < HID; ++k) {
    float xv = xr[k];
    cl0 += xv * cw[0 * HID + k];
    cl1 += xv * cw[1 * HID + k];
    cl2 += xv * cw[2 * HID + k];
    cl3 += xv * cw[3 * HID + k];
  }
  float m  = fmaxf(fmaxf(cl0, cl1), fmaxf(cl2, cl3));
  float se = __expf(cl0 - m) + __expf(cl1 - m) + __expf(cl2 - m) + __expf(cl3 - m);
  float lse = m + __logf(se);
  int   cid = cluster_of(y[n]);
  float clc = (cid == 0) ? cl0 : (cid == 1) ? cl1 : (cid == 2) ? cl2 : cl3;
  float cluster_ll = clc - lse;
  float logZ = __logf(sumexp[n * 4 + cid]);
  float lp_y = tgt[n] - logZ;
  nll[n] = -(cluster_ll + lp_y);
}

// ---------------------------------------------------------------- launch
extern "C" void kernel_launch(void* const* d_in, const int* in_sizes, int n_in,
                              void* d_out, int out_size, void* d_ws, size_t ws_size,
                              hipStream_t stream) {
  const float* x  = (const float*)d_in[0];
  const int*   y  = (const int*)d_in[1];
  const float* cw = (const float*)d_in[2];
  const float* W  = (const float*)d_in[3];
  float* nll = (float*)d_out;

  char* ws = (char*)d_ws;
  __bf16* xb     = (__bf16*)ws;                               // 8 MB
  float*  sumexp = (float*)(ws + (size_t)NTOK * HID * 2);     // 64 KB
  float*  tgt    = sumexp + NTOK * 4;                         // 16 KB

  prep_kernel<<<(NTOK * HID + 255) / 256, 256, 0, stream>>>(x, xb, sumexp);
  tgt_kernel<<<NTOK, 128, 0, stream>>>(x, y, W, tgt);
  gemm_kernel<<<NBLK, 128, 0, stream>>>(xb, W, sumexp);
  final_kernel<<<(NTOK + 127) / 128, 128, 0, stream>>>(x, y, cw, sumexp, tgt, nll);
}
